// TpsGridGen_30202210025942
// MI455X (gfx1250) — matrix operand order repaired and checked by reading, not compiled, not run on hardware
//
#include <hip/hip_runtime.h>

// TPS grid generator for MI455X (gfx1250, wave32).
//
//   out[b,h,w,d,c] = Ac[b,c,0] + grid·Ac[b,c,1:4] + sum_n U(p,n)·Wc[b,c,n]
//
// Skinny GEMM: F[p,0:32] = [r2*log2(r2) vs 27 ctrl pts, 1, x, y, z, 0] (f16),
// B = Wfull[0:32, n], n = b*3+c (6 real columns of 16); ln2 folded into the
// 27 RBF rows of B so the hot loop uses raw v_log_f32 (log2).
// One v_wmma_f32_16x16x32_f16 per 16 grid points yields all 6 outputs.

#define HWD   (128 * 128 * 128)
#define NTILE (HWD / 16)          // 131072 tiles of 16 points

typedef __attribute__((ext_vector_type(16))) _Float16 v16h;
typedef __attribute__((ext_vector_type(8)))  float    v8f;

// ---------------------------------------------------------------------------
// Prep: Wfull[n][k] = sum_m Li[k][m] * theta[b*81 + c*27 + m],  n = b*3+c.
// Rows 0..26 (RBF) are pre-scaled by ln(2); rows 27..30 affine; row 31 = 0.
// ---------------------------------------------------------------------------
__global__ void tps_prep_kernel(const float* __restrict__ theta,
                                const float* __restrict__ Li,
                                float* __restrict__ Wfull) {
    int t = threadIdx.x;
    if (t >= 192) return;
    int n = t >> 5;      // 0..5  (b*3+c)
    int k = t & 31;      // 0..31
    float s = 0.f;
    if (k < 31) {
        int b = n / 3, c = n % 3;
        const float* q = theta + b * 81 + c * 27;
        for (int m = 0; m < 27; ++m) s += Li[k * 31 + m] * q[m];
    }
    if (k < 27) s *= 0.69314718055994530942f;   // fold ln2: U = ln2*r2*log2(r2)
    Wfull[n * 32 + k] = s;
}

// ---------------------------------------------------------------------------
// Main: one wave = 16 grid points per WMMA tile, grid-stride over tiles.
// ---------------------------------------------------------------------------
__global__ void tps_main_kernel(const float* __restrict__ grid,
                                const float* __restrict__ P,
                                const float* __restrict__ Wfull,
                                float* __restrict__ out) {
    const int lane   = threadIdx.x & 31;
    const int wave   = (int)((blockIdx.x * blockDim.x + threadIdx.x) >> 5);
    const int nwaves = (int)((gridDim.x * blockDim.x) >> 5);
    const int half   = lane >> 4;     // lane group (0: lanes 0-15, 1: 16-31)
    const int n      = lane & 15;     // A row within tile / D column

    // --- Per-lane A-fragment K indices (ISA 16-bit A 16x32 layout):
    //     k(j) = (j<8 ? j : j+8) + 8*half.
    //     Slots 0..10 are RBF features for BOTH halves; slots 11..15 are RBF
    //     for half==0 (k=19..23) and affine {1,x,y,z,0} for half==1 (k=27..31).
    //     r2 = s - 2*p·g + |p|^2 with s = x^2+y^2+z^2 per tile.
    float m2x[16], m2y[16], m2z[16], q[16];
    #pragma unroll
    for (int j = 0; j < 16; ++j) {
        int k = ((j < 8) ? j : j + 8) + (half ? 8 : 0);
        float px = 0.f, py = 0.f, pz = 0.f;
        if (k < 27) {
            px = P[k * 3 + 0];
            py = P[k * 3 + 1];
            pz = P[k * 3 + 2];
        }
        m2x[j] = -2.f * px;
        m2y[j] = -2.f * py;
        m2z[j] = -2.f * pz;
        q[j]   = px * px + py * py + pz * pz;
    }

    // --- B-fragment (loop invariant): k(j) = (j<8 ? j : j+8) + 8*half,
    //     matching the A layout; column N = lane%16, only N<6 real ---
    v16h bfrag;
    #pragma unroll
    for (int j = 0; j < 16; ++j) {
        int kb = ((j < 8) ? j : j + 8) + (half ? 8 : 0);
        float wv = (n < 6 && kb < 31) ? Wfull[n * 32 + kb] : 0.f;
        bfrag[j] = (_Float16)wv;
    }

    // --- Per-lane output base (valid only when n < 6) ---
    const int m0 = half * 8;                        // D rows m0..m0+7
    const int bb = n / 3, cc = n % 3;
    float* const obase = out + (size_t)bb * (size_t)(HWD * 3) + cc + (size_t)m0 * 3;
    const bool writer = (n < 6);
    const bool aff    = (half != 0);

    #pragma unroll 2
    for (int tile = wave; tile < NTILE; tile += nwaves) {
        // This lane's A-row grid point
        const int p = tile * 16 + n;
        const float x = grid[p * 3 + 0];
        const float y = grid[p * 3 + 1];
        const float z = grid[p * 3 + 2];
        const float s = __builtin_fmaf(z, z, __builtin_fmaf(y, y, x * x));

        // Branch-free feature build: U = r2*log2(r2), clamp avoids 0*(-inf)
        float fv[16];
        #pragma unroll
        for (int j = 0; j < 16; ++j) {
            float r2 = __builtin_fmaf(m2x[j], x,
                       __builtin_fmaf(m2y[j], y,
                       __builtin_fmaf(m2z[j], z, s + q[j])));
            float r2c = __builtin_fmaxf(r2, 1e-37f);   // r2==0 -> U ~ 1e-35 -> f16 0
            fv[j] = r2c * __builtin_amdgcn_logf(r2c);  // v_log_f32 (log2)
        }
        // Patch affine slots (half==1 only): k = 27,28,29,30,31
        fv[11] = aff ? 1.f : fv[11];
        fv[12] = aff ? x   : fv[12];
        fv[13] = aff ? y   : fv[13];
        fv[14] = aff ? z   : fv[14];
        fv[15] = aff ? 0.f : fv[15];

        v16h afrag;
        #pragma unroll
        for (int j = 0; j < 16; ++j) afrag[j] = (_Float16)fv[j];

        // D = A x B + 0   (EXEC all-1s: uniform control flow in the loop)
        v8f acc = {};
        acc = __builtin_amdgcn_wmma_f32_16x16x32_f16(
            /*neg_a=*/false, afrag, /*neg_b=*/false, bfrag,
            /*c_mod=*/(short)0, acc, /*reuse_a=*/false, /*reuse_b=*/false);

        // Scatter: lane covers column n=(b*3+c); VGPR r covers row m0+r
        if (writer) {
            float* o = obase + (size_t)tile * 48;   // 16 points * 3 channels
            #pragma unroll
            for (int r = 0; r < 8; ++r) o[r * 3] = acc[r];
        }
    }
}

extern "C" void kernel_launch(void* const* d_in, const int* in_sizes, int n_in,
                              void* d_out, int out_size, void* d_ws, size_t ws_size,
                              hipStream_t stream) {
    const float* theta = (const float*)d_in[0];   // (B, 81)
    const float* grid  = (const float*)d_in[1];   // (128,128,128,3)
    const float* P     = (const float*)d_in[2];   // (27, 3)
    const float* Li    = (const float*)d_in[3];   // (31, 31)
    float* Wfull = (float*)d_ws;                  // 6 x 32 f32 = 768 B
    float* out   = (float*)d_out;                 // (2,128,128,128,3)

    tps_prep_kernel<<<1, 192, 0, stream>>>(theta, Li, Wfull);
    // 1024 blocks x 256 threads = 8192 waves; 16 tiles (256 points) per wave
    tps_main_kernel<<<1024, 256, 0, stream>>>(grid, P, Wfull, out);
}
